// TemporalLSTM_36498632081924
// MI455X (gfx1250) — matrix operand order repaired and checked by reading, not compiled, not run on hardware
//
#include <hip/hip_runtime.h>

#define B_SZ 4096
#define T_SZ 48
#define FIN  225
#define H_SZ 128
#define NCLS 100
#define KPAD 256          // F_IN padded to multiple of 32
#define G4H  512          // 4*H

typedef __bf16 bf16_t;
typedef bf16_t v16bf __attribute__((ext_vector_type(16)));
typedef float  v8f   __attribute__((ext_vector_type(8)));
typedef unsigned int u32x4 __attribute__((ext_vector_type(4)));

union ABReg { v16bf v; u32x4 q[2]; };

__device__ __forceinline__ unsigned short f2bf(float f) {
    unsigned int u = __float_as_uint(f);
    u += 0x7FFFu + ((u >> 16) & 1u);          // round-to-nearest-even
    return (unsigned short)(u >> 16);
}
// Fast activations on the hardware transcendental pipe (co-executes with WMMA).
__device__ __forceinline__ float sigm(float x) {
    // 1/(1+e^-x); exp2(-inf)=0 -> 1, exp2(+inf)=inf -> rcp=0. Saturates right.
    return __builtin_amdgcn_rcpf(1.0f + __builtin_amdgcn_exp2f(-1.4426950408889634f * x));
}
__device__ __forceinline__ float tanh_fast(float x) {
    // tanh(x) = 1 - 2/(1+e^{2x})
    return 1.0f - 2.0f * __builtin_amdgcn_rcpf(
        1.0f + __builtin_amdgcn_exp2f(2.8853900817779268f * x));
}

// ---------------------------------------------------------------------------
// Prep: fp32 weights -> bf16 (w_ih zero-padded K 225->256), biases folded.
// ---------------------------------------------------------------------------
__global__ void prep_kernel(
    const float* __restrict__ wihf, const float* __restrict__ whhf,
    const float* __restrict__ bihf, const float* __restrict__ bhhf,
    const float* __restrict__ wihb, const float* __restrict__ whhb,
    const float* __restrict__ bihb, const float* __restrict__ bhhb,
    unsigned short* __restrict__ wih_bf,   // [2][512][256]
    unsigned short* __restrict__ whh_bf,   // [2][512][128]
    float* __restrict__ bias)              // [2][512]
{
    const int NW_IH = 2 * G4H * KPAD;      // 262144
    const int NW_HH = 2 * G4H * H_SZ;      // 131072
    const int NB    = 2 * G4H;             // 1024
    int idx = blockIdx.x * 256 + threadIdx.x;
    if (idx < NW_IH) {
        int d   = idx / (G4H * KPAD);
        int rem = idx % (G4H * KPAD);
        int r = rem / KPAD, k = rem % KPAD;
        const float* w = d ? wihb : wihf;
        wih_bf[idx] = (k < FIN) ? f2bf(w[r * FIN + k]) : (unsigned short)0;
    } else if (idx < NW_IH + NW_HH) {
        int j = idx - NW_IH;
        int d = j / (G4H * H_SZ);
        int rem = j % (G4H * H_SZ);
        const float* w = d ? whhb : whhf;
        whh_bf[j] = f2bf(w[rem]);
    } else if (idx < NW_IH + NW_HH + NB) {
        int j = idx - NW_IH - NW_HH;
        int d = j / G4H, r = j % G4H;
        bias[j] = d ? (bihb[r] + bhhb[r]) : (bihf[r] + bhhf[r]);
    }
}

// ---------------------------------------------------------------------------
// One-time x fp32 -> bf16 with K padding 225->256. [4096][48][256] bf16.
// ---------------------------------------------------------------------------
__global__ __launch_bounds__(256) void xconv_kernel(
    const float* __restrict__ x, unsigned short* __restrict__ xbf)
{
    size_t idx = (size_t)blockIdx.x * 256 + threadIdx.x;  // over B*T*256
    int k = (int)(idx & (KPAD - 1));
    size_t bt = idx >> 8;
    float v = 0.0f;
    if (k < FIN) v = x[bt * FIN + k];
    xbf[idx] = f2bf(v);
}

// ---------------------------------------------------------------------------
// Fused bidirectional LSTM scan + mean pool. One block = 32 batch rows,
// one direction, all 48 timesteps. 8 waves: 2(M-tiles) x 4(N-slices of H).
// ---------------------------------------------------------------------------
__global__ __launch_bounds__(256) void lstm_scan_kernel(
    const float* __restrict__ x,                 // [4096][48][225] fp32
    const unsigned short* __restrict__ xbf,      // [4096][48][256] bf16 (or null)
    const unsigned short* __restrict__ wih_bf,   // [2][512][256] bf16
    const unsigned short* __restrict__ whh_bf,   // [2][512][128] bf16
    const float* __restrict__ bias,              // [2][512]
    float* __restrict__ pooled)                  // [4096][256] fp32
{
    __shared__ __align__(16) unsigned short xs[32][KPAD];   // x tile, bf16
    __shared__ __align__(16) unsigned short hsh[32][H_SZ];  // h state, bf16

    const int tid  = threadIdx.x;
    const int lane = tid & 31;
    const int wv   = tid >> 5;            // 0..7
    const int mw   = wv >> 2;             // 0..1  (M tile)
    const int nw   = wv & 3;              // 0..3  (N slice of H)
    const int half = (lane >> 4) & 1;
    const int lcol = lane & 15;
    const int m0   = mw * 16;
    const int dir   = blockIdx.y;
    const int bbase = blockIdx.x * 32;

    const unsigned short* wih = wih_bf + (size_t)dir * G4H * KPAD;
    const unsigned short* whh = whh_bf + (size_t)dir * G4H * H_SZ;
    const float*          bb  = bias + dir * G4H;

    // Per-accumulator bias (constant over rows): column = g*128 + nw*32 + t2*16 + lcol
    float bgt[4][2];
#pragma unroll
    for (int g = 0; g < 4; ++g)
#pragma unroll
        for (int t2 = 0; t2 < 2; ++t2)
            bgt[g][t2] = bb[g * H_SZ + nw * 32 + t2 * 16 + lcol];

    v8f cst[2], hsum[2];
#pragma unroll
    for (int t2 = 0; t2 < 2; ++t2)
#pragma unroll
        for (int v = 0; v < 8; ++v) { cst[t2][v] = 0.0f; hsum[t2][v] = 0.0f; }

    // h(0) = 0
    for (int i = tid; i < 32 * H_SZ; i += 256)
        ((unsigned short*)hsh)[i] = 0;
    // zero the padded columns of xs once (fallback path never rewrites them)
    if (tid >= FIN) {
#pragma unroll
        for (int r = 0; r < 32; ++r) xs[r][tid] = 0;
    }

    for (int s = 0; s < T_SZ; ++s) {
        const int t = dir ? (T_SZ - 1 - s) : s;

        // Stage x tile into LDS.
        if (xbf) {
            // bf16 path: pure 16B copies, no conversion VALU.
#pragma unroll
            for (int i = 0; i < 4; ++i) {
                int c = tid + i * 256;          // 0..1023 chunks of 8 shorts
                int r = c >> 5;
                int o = (c & 31) * 8;
                *(u32x4*)&xs[r][o] =
                    *(const u32x4*)(xbf + ((size_t)(bbase + r) * T_SZ + t) * KPAD + o);
            }
        } else if (tid < FIN) {
            const float* xp = x + ((size_t)bbase * T_SZ + t) * FIN + tid;
#pragma unroll 4
            for (int r = 0; r < 32; ++r)
                xs[r][tid] = f2bf(xp[(size_t)r * (T_SZ * FIN)]);
        }
        __syncthreads();   // xs ready; prev-step hsh writes visible

        // Opaque memory clobber: stops LICM from hoisting the (loop-invariant)
        // weight loads out of the scan, which would spill ~768 VGPRs to scratch.
        asm volatile("" ::: "memory");

        // gates(pre-bias) = x_t @ w_ih^T + h @ w_hh^T ; bias folded in pointwise
        v8f acc[4][2];

        // x contribution, k-step 0 peeled with zero C (inline-0 SRC2)
        {
            ABReg a;
            const unsigned short* ap = &xs[m0 + lcol][half * 8];
            a.q[0] = *(const u32x4*)ap;
            a.q[1] = *(const u32x4*)(ap + 16);
            const v8f zc = {};
#pragma unroll
            for (int g = 0; g < 4; ++g)
#pragma unroll
                for (int t2 = 0; t2 < 2; ++t2) {
                    ABReg bm;
                    const unsigned short* bp =
                        wih + (size_t)(g * H_SZ + nw * 32 + t2 * 16 + lcol) * KPAD
                            + half * 8;
                    bm.q[0] = *(const u32x4*)bp;
                    bm.q[1] = *(const u32x4*)(bp + 16);
                    acc[g][t2] = __builtin_amdgcn_wmma_f32_16x16x32_bf16(
                        false, a.v, false, bm.v, (short)0, zc, false, false);
                }
        }
        // x contribution: remaining k-steps (K padded to 256)
        for (int ks = 1; ks < 8; ++ks) {
            ABReg a;
            const unsigned short* ap = &xs[m0 + lcol][ks * 32 + half * 8];
            a.q[0] = *(const u32x4*)ap;
            a.q[1] = *(const u32x4*)(ap + 16);
#pragma unroll
            for (int g = 0; g < 4; ++g)
#pragma unroll
                for (int t2 = 0; t2 < 2; ++t2) {
                    ABReg bm;
                    const unsigned short* bp =
                        wih + (size_t)(g * H_SZ + nw * 32 + t2 * 16 + lcol) * KPAD
                            + ks * 32 + half * 8;
                    bm.q[0] = *(const u32x4*)bp;
                    bm.q[1] = *(const u32x4*)(bp + 16);
                    acc[g][t2] = __builtin_amdgcn_wmma_f32_16x16x32_bf16(
                        false, a.v, false, bm.v, (short)0, acc[g][t2], false, false);
                }
        }
        // h contribution: K = 128, 4 k-steps of 32
        for (int ks = 0; ks < 4; ++ks) {
            ABReg a;
            const unsigned short* ap = &hsh[m0 + lcol][ks * 32 + half * 8];
            a.q[0] = *(const u32x4*)ap;
            a.q[1] = *(const u32x4*)(ap + 16);
#pragma unroll
            for (int g = 0; g < 4; ++g)
#pragma unroll
                for (int t2 = 0; t2 < 2; ++t2) {
                    ABReg bm;
                    const unsigned short* bp =
                        whh + (size_t)(g * H_SZ + nw * 32 + t2 * 16 + lcol) * H_SZ
                            + ks * 32 + half * 8;
                    bm.q[0] = *(const u32x4*)bp;
                    bm.q[1] = *(const u32x4*)(bp + 16);
                    acc[g][t2] = __builtin_amdgcn_wmma_f32_16x16x32_bf16(
                        false, a.v, false, bm.v, (short)0, acc[g][t2], false, false);
                }
        }
        __syncthreads();   // all waves done reading hsh before overwrite

        // Pointwise LSTM cell update; c stays fp32 in registers.
        // C/D layout: element (row = m0 + 8*half + v, col = nw*32 + t2*16 + lcol)
#pragma unroll
        for (int t2 = 0; t2 < 2; ++t2)
#pragma unroll
            for (int v = 0; v < 8; ++v) {
                float iv = sigm(acc[0][t2][v] + bgt[0][t2]);
                float fv = sigm(acc[1][t2][v] + bgt[1][t2]);
                float gv = tanh_fast(acc[2][t2][v] + bgt[2][t2]);
                float ov = sigm(acc[3][t2][v] + bgt[3][t2]);
                float cv = fv * cst[t2][v] + iv * gv;
                cst[t2][v] = cv;
                float hv = ov * tanh_fast(cv);
                hsum[t2][v] += hv;
                hsh[m0 + half * 8 + v][nw * 32 + t2 * 16 + lcol] = f2bf(hv);
            }
    }

    // mean over T; fwd -> cols [0,128), bwd -> cols [128,256)
#pragma unroll
    for (int t2 = 0; t2 < 2; ++t2)
#pragma unroll
        for (int v = 0; v < 8; ++v)
            pooled[(size_t)(bbase + m0 + half * 8 + v) * 256
                   + dir * H_SZ + nw * 32 + t2 * 16 + lcol]
                = hsum[t2][v] * (1.0f / (float)T_SZ);
}

// ---------------------------------------------------------------------------
// Fused FC head: relu(pooled @ fc1^T + b1) @ fc2^T + b2. One block per row.
// ---------------------------------------------------------------------------
__global__ __launch_bounds__(128) void fc_head_kernel(
    const float* __restrict__ pooled,   // [4096][256]
    const float* __restrict__ fc1w,     // [128][256]
    const float* __restrict__ fc1b,     // [128]
    const float* __restrict__ fc2w,     // [100][128]
    const float* __restrict__ fc2b,     // [100]
    float* __restrict__ out)            // [4096][100]
{
    __shared__ float row[256];
    __shared__ float h1[128];
    const int b = blockIdx.x, tid = threadIdx.x;
    row[tid]       = pooled[(size_t)b * 256 + tid];
    row[tid + 128] = pooled[(size_t)b * 256 + 128 + tid];
    __syncthreads();
    float a = fc1b[tid];
    const float* w1 = fc1w + (size_t)tid * 256;
#pragma unroll 8
    for (int k = 0; k < 256; ++k) a = fmaf(row[k], w1[k], a);
    h1[tid] = fmaxf(a, 0.0f);
    __syncthreads();
    if (tid < NCLS) {
        float a2 = fc2b[tid];
        const float* w2 = fc2w + (size_t)tid * H_SZ;
#pragma unroll 8
        for (int k = 0; k < H_SZ; ++k) a2 = fmaf(h1[k], w2[k], a2);
        out[(size_t)b * NCLS + tid] = a2;
    }
}

// ---------------------------------------------------------------------------
extern "C" void kernel_launch(void* const* d_in, const int* in_sizes, int n_in,
                              void* d_out, int out_size, void* d_ws, size_t ws_size,
                              hipStream_t stream) {
    const float* x    = (const float*)d_in[0];
    const float* wihf = (const float*)d_in[1];
    const float* whhf = (const float*)d_in[2];
    const float* bihf = (const float*)d_in[3];
    const float* bhhf = (const float*)d_in[4];
    const float* wihb = (const float*)d_in[5];
    const float* whhb = (const float*)d_in[6];
    const float* bihb = (const float*)d_in[7];
    const float* bhhb = (const float*)d_in[8];
    const float* fc1w = (const float*)d_in[9];
    const float* fc1b = (const float*)d_in[10];
    const float* fc2w = (const float*)d_in[11];
    const float* fc2b = (const float*)d_in[12];
    float* out = (float*)d_out;

    // Workspace layout (all 256B-aligned):
    char* ws = (char*)d_ws;
    const size_t OFF_WIH  = 0;                         // 524288 B
    const size_t OFF_WHH  = 524288;                    // 262144 B
    const size_t OFF_BIAS = 524288 + 262144;           //   4096 B
    const size_t OFF_POOL = OFF_BIAS + 4096;           // 4194304 B
    const size_t OFF_XBF  = OFF_POOL + 4194304;        // 100663296 B
    const size_t XBF_BYTES = (size_t)B_SZ * T_SZ * KPAD * sizeof(unsigned short);

    unsigned short* wih_bf = (unsigned short*)(ws + OFF_WIH);
    unsigned short* whh_bf = (unsigned short*)(ws + OFF_WHH);
    float*          biasws = (float*)(ws + OFF_BIAS);
    float*          pooled = (float*)(ws + OFF_POOL);
    const bool use_xbf = ws_size >= OFF_XBF + XBF_BYTES;
    unsigned short* xbf = use_xbf ? (unsigned short*)(ws + OFF_XBF) : nullptr;

    prep_kernel<<<(2 * G4H * KPAD + 2 * G4H * H_SZ + 2 * G4H + 255) / 256, 256, 0, stream>>>(
        wihf, whhf, bihf, bhhf, wihb, whhb, bihb, bhhb, wih_bf, whh_bf, biasws);

    if (use_xbf)
        xconv_kernel<<<B_SZ * T_SZ, 256, 0, stream>>>(x, xbf);

    dim3 grid(B_SZ / 32, 2);
    lstm_scan_kernel<<<grid, 256, 0, stream>>>(x, xbf, wih_bf, whh_bf, biasws, pooled);

    fc_head_kernel<<<B_SZ, 128, 0, stream>>>(pooled, fc1w, fc1b, fc2w, fc2b, out);
}
